// TransformerLaneCond_32950989095611
// MI455X (gfx1250) — compile-verified, matrix operand order checked by reading, not run on hardware
//
#include <hip/hip_runtime.h>

// ---------------- model constants ----------------
#define BB 8192
#define TT 11
#define LL 16
#define NN 32
#define FF 26
#define DD 128
#define NHH 4
#define HDD 32
#define DFFF 256
#define NBHH 64
#define FUTT 30

typedef __attribute__((ext_vector_type(16))) _Float16 v16h;
typedef __attribute__((ext_vector_type(8)))  float    v8f;

#define GEMM_RELU  1
#define GEMM_ACCUM 2
#define GEMM_MASK  4

#define ASP 264   // LDS pitch in halves for the A block (bank-conflict free, 16B-aligned rows)

union FragA { v16h h; uint u[8]; };
union Pack8 { _Float16 h[8]; uint4 u; };

// =====================================================================
// WMMA GEMM: Out[M x N] = A[M x K] @ W[K x N] (+bias)(+Out)(relu)(rowmask)
// Block: 256 threads (8 waves). Block tile: 64 rows x 128 cols.
// Wave w owns cols [blockIdx.y*128 + w*16, +16) across four 16-row tiles.
// Requires M % 64 == 0, K <= 256. GUARD=false requires K%32==0, N%128==0.
// Whole 64xK A-block staged to LDS once (f16); single barrier; k-loop runs
// from LDS (ds_load_b128 pairs) + direct W loads (L2-resident weights).
// =====================================================================
template<bool GUARD>
__global__ __launch_bounds__(256)
void wmma_gemm_kernel(const float* __restrict__ A, int lda,
                      const float* __restrict__ W, int ldw,
                      const float* __restrict__ bias,
                      float* __restrict__ Out, int ldo,
                      int M, int K, int N, int flags,
                      const float* __restrict__ rowmask)
{
    __shared__ __align__(16) _Float16 As[64][ASP];

    const int lane = threadIdx.x & 31;
    const int wid  = threadIdx.x >> 5;
    const int m0   = blockIdx.x * 64;
    const int col0 = blockIdx.y * 128 + wid * 16;
    const int hf   = lane >> 4;
    const int mrow = lane & 15;
    const bool cok = GUARD ? (col0 + mrow < N) : true;
    const int  gc  = GUARD ? min(col0 + mrow, N - 1) : (col0 + mrow);
    const float* Wc = W + gc;
    const int nk = (K + 31) >> 5;

    // ---- stage the whole 64 x K A block into LDS (one shot) ----
    if (GUARD) {
        const int kpad = nk << 5;
        for (int e = threadIdx.x; e < 64 * kpad; e += 256) {
            int r = e / kpad, c = e % kpad;
            int gk = min(c, K - 1);
            float v = A[(long)(m0 + r) * lda + gk];
            if (c >= K) v = 0.f;
            As[r][c] = (_Float16)v;
        }
    } else {
        const int cpr = K >> 3;                 // 8-float chunks per row
        for (int ch = threadIdx.x; ch < 64 * cpr; ch += 256) {
            int r = ch / cpr;
            int c = (ch % cpr) * 8;
            const float4* src = reinterpret_cast<const float4*>(A + (long)(m0 + r) * lda + c);
            float4 f0 = src[0];
            float4 f1 = src[1];
            Pack8 pk;
            pk.h[0] = (_Float16)f0.x; pk.h[1] = (_Float16)f0.y;
            pk.h[2] = (_Float16)f0.z; pk.h[3] = (_Float16)f0.w;
            pk.h[4] = (_Float16)f1.x; pk.h[5] = (_Float16)f1.y;
            pk.h[6] = (_Float16)f1.z; pk.h[7] = (_Float16)f1.w;
            *reinterpret_cast<uint4*>(&As[r][c]) = pk.u;
        }
    }
    __syncthreads();

    // ---- k-loop: LDS fragments + direct W fragments, no barriers ----
    v8f acc[4] = {};
    for (int kk = 0; kk < nk; ++kk) {
        const int k0 = kk << 5;

        v16h bf;
#pragma unroll
        for (int j = 0; j < 16; ++j) {
            int kl = (j & 7) + (hf << 3) + ((j >> 3) << 4);
            float wv;
            if (GUARD) {
                int gk = min(k0 + kl, K - 1);
                wv = Wc[(long)gk * ldw];
                if (k0 + kl >= K || !cok) wv = 0.f;
            } else {
                wv = Wc[(long)(k0 + kl) * ldw];
            }
            bf[j] = (_Float16)wv;
        }

#pragma unroll
        for (int t = 0; t < 4; ++t) {
            FragA af;
#pragma unroll
            for (int jj = 0; jj < 8; ++jj) {
                int kl = k0 + 2 * (jj & 3) + (hf << 3) + ((jj >> 2) << 4);  // even
                af.u[jj] = *reinterpret_cast<const uint*>(&As[t * 16 + mrow][kl]);
            }
            acc[t] = __builtin_amdgcn_wmma_f32_16x16x32_f16(false, af.h, false, bf,
                                                            (short)0, acc[t], false, false);
        }
    }

    // ---- epilogue: batch all auxiliary loads, then combine + store ----
    float bv = 0.f;
    if (bias && cok) bv = bias[gc];

    float prev[4][8];
    if (flags & GEMM_ACCUM) {
#pragma unroll
        for (int t = 0; t < 4; ++t)
#pragma unroll
            for (int r = 0; r < 8; ++r) {
                int gm = m0 + t * 16 + r + (hf << 3);
                prev[t][r] = cok ? Out[(long)gm * ldo + gc] : 0.f;
            }
    }
    float rmk[4][8];
    if (flags & GEMM_MASK) {
#pragma unroll
        for (int t = 0; t < 4; ++t)
#pragma unroll
            for (int r = 0; r < 8; ++r) {
                int gm = m0 + t * 16 + r + (hf << 3);
                rmk[t][r] = rowmask[gm];
            }
    }

#pragma unroll
    for (int t = 0; t < 4; ++t) {
#pragma unroll
        for (int r = 0; r < 8; ++r) {
            int gm = m0 + t * 16 + r + (hf << 3);
            float v = acc[t][r];
            if (flags & GEMM_ACCUM) v += prev[t][r];
            v += bv;
            if (flags & GEMM_RELU) v = fmaxf(v, 0.f);
            if (flags & GEMM_MASK) v *= rmk[t][r];
            if (cok) Out[(long)gm * ldo + gc] = v;
        }
    }
}

// =====================================================================
// Fused attention per (batch, head), one wave each (8 waves / block):
// S = Q K^T * scale (1 WMMA), masked softmax across N-lanes, O = P V (2 WMMA).
// Q: [B,TQ,128], K/V: [B,TK,128], head h uses cols [h*32, h*32+32).
// Padded rows/keys use clamped (duplicate) loads; the softmax mask and the
// zeroed P tail make them mathematically inert, so no per-element selects.
// =====================================================================
template<int TQi, int TKi, bool MASKED>
__global__ __launch_bounds__(256)
void wmma_attn_kernel(const float* __restrict__ Qp, const float* __restrict__ Kp,
                      const float* __restrict__ Vp, float* __restrict__ Op,
                      const float* __restrict__ maskp, float scale)
{
    __shared__ float Pb[8][16][17];

    const int lane = threadIdx.x & 31;
    const int wid  = threadIdx.x >> 5;
    const int gid  = blockIdx.x * 8 + wid;
    const int b    = gid >> 2;          // NH = 4
    const int h    = gid & 3;
    const int hf   = lane >> 4;
    const int mrow = lane & 15;

    const float* Qb = Qp + (long)b * TQi * DD + h * HDD;
    const float* Kb = Kp + (long)b * TKi * DD + h * HDD;
    const float* Vb = Vp + (long)b * TKi * DD + h * HDD;

    const int rq = (TQi < 16) ? min(mrow, TQi - 1) : mrow;
    const int rk = (TKi < 16) ? min(mrow, TKi - 1) : mrow;

    // Q/K fragments: two 8-float runs per lane -> 4x float4 each
    const float* qrow = Qb + rq * DD + (hf << 3);
    const float* krow = Kb + rk * DD + (hf << 3);
    float4 q0 = *(const float4*)(qrow);
    float4 q1 = *(const float4*)(qrow + 4);
    float4 q2 = *(const float4*)(qrow + 16);
    float4 q3 = *(const float4*)(qrow + 20);
    float4 k0 = *(const float4*)(krow);
    float4 k1 = *(const float4*)(krow + 4);
    float4 k2 = *(const float4*)(krow + 16);
    float4 k3 = *(const float4*)(krow + 20);

    v16h qf, kf;
    qf[0]=(_Float16)q0.x; qf[1]=(_Float16)q0.y; qf[2]=(_Float16)q0.z; qf[3]=(_Float16)q0.w;
    qf[4]=(_Float16)q1.x; qf[5]=(_Float16)q1.y; qf[6]=(_Float16)q1.z; qf[7]=(_Float16)q1.w;
    qf[8]=(_Float16)q2.x; qf[9]=(_Float16)q2.y; qf[10]=(_Float16)q2.z; qf[11]=(_Float16)q2.w;
    qf[12]=(_Float16)q3.x; qf[13]=(_Float16)q3.y; qf[14]=(_Float16)q3.z; qf[15]=(_Float16)q3.w;
    kf[0]=(_Float16)k0.x; kf[1]=(_Float16)k0.y; kf[2]=(_Float16)k0.z; kf[3]=(_Float16)k0.w;
    kf[4]=(_Float16)k1.x; kf[5]=(_Float16)k1.y; kf[6]=(_Float16)k1.z; kf[7]=(_Float16)k1.w;
    kf[8]=(_Float16)k2.x; kf[9]=(_Float16)k2.y; kf[10]=(_Float16)k2.z; kf[11]=(_Float16)k2.w;
    kf[12]=(_Float16)k3.x; kf[13]=(_Float16)k3.y; kf[14]=(_Float16)k3.z; kf[15]=(_Float16)k3.w;

    v8f s = {};
    s = __builtin_amdgcn_wmma_f32_16x16x32_f16(false, qf, false, kf, (short)0, s, false, false);

    bool valid = (mrow < TKi);
    if (MASKED) valid = valid && (maskp[(long)b * TKi + mrow] >= 0.5f);

    float p[8];
#pragma unroll
    for (int r = 0; r < 8; ++r) {
        float sv = valid ? s[r] * scale : -1e9f;
        float mx = sv;
        mx = fmaxf(mx, __shfl_xor(mx, 1));
        mx = fmaxf(mx, __shfl_xor(mx, 2));
        mx = fmaxf(mx, __shfl_xor(mx, 4));
        mx = fmaxf(mx, __shfl_xor(mx, 8));
        float e = __expf(sv - mx);
        float su = e;
        su += __shfl_xor(su, 1);
        su += __shfl_xor(su, 2);
        su += __shfl_xor(su, 4);
        su += __shfl_xor(su, 8);
        p[r] = e / su;
    }
#pragma unroll
    for (int r = 0; r < 8; ++r)
        Pb[wid][r + (hf << 3)][mrow] = p[r];   // P[query][key]
    __syncthreads();

    v16h pf;
#pragma unroll
    for (int j = 0; j < 16; ++j) {
        int kd = (j & 7) + (hf << 3) + ((j >> 3) << 4);   // key index 0..31
        float pv = (kd < 16) ? Pb[wid][mrow][kd] : 0.f;
        pf[j] = (_Float16)pv;
    }
#pragma unroll
    for (int c = 0; c < 2; ++c) {
        v16h vf;
#pragma unroll
        for (int j = 0; j < 16; ++j) {
            int key = (j & 7) + (hf << 3) + ((j >> 3) << 4);
            int kc  = min(key, TKi - 1);        // clamped; p==0 for padded keys
            vf[j] = (_Float16)Vb[kc * DD + c * 16 + mrow];
        }
        v8f o = {};
        o = __builtin_amdgcn_wmma_f32_16x16x32_f16(false, pf, false, vf, (short)0, o, false, false);
#pragma unroll
        for (int r = 0; r < 8; ++r) {
            int q = r + (hf << 3);
            if (q < TQi)
                Op[((long)b * TQi + q) * DD + h * HDD + c * 16 + mrow] = o[r];
        }
    }
}

// ---------------- small VALU kernels ----------------

__global__ __launch_bounds__(128)
void embed_kernel(const float* __restrict__ sdc, const float* __restrict__ w,
                  const float* __restrict__ bb, const float* __restrict__ pos,
                  float* __restrict__ X)
{
    int bt = blockIdx.x, d = threadIdx.x;
    int t = bt % TT;
    float a0 = sdc[(long)bt * 2], a1 = sdc[(long)bt * 2 + 1];
    X[(long)bt * DD + d] = a0 * w[d] + a1 * w[DD + d] + bb[d] + pos[t * DD + d];
}

// out = flag ? x : LN(x + r)
__global__ __launch_bounds__(128)
void add_ln_kernel(const float* __restrict__ x, const float* __restrict__ r,
                   const float* __restrict__ g, const float* __restrict__ bb,
                   float* __restrict__ out, const float* __restrict__ flg, int flgdiv)
{
    __shared__ float red[128];
    int row = blockIdx.x, d = threadIdx.x;
    float xv = x[(long)row * DD + d];
    float v = xv + r[(long)row * DD + d];
    red[d] = v; __syncthreads();
    for (int s = 64; s > 0; s >>= 1) { if (d < s) red[d] += red[d + s]; __syncthreads(); }
    float mu = red[0] * (1.f / DD);
    __syncthreads();
    float dv = v - mu;
    red[d] = dv * dv; __syncthreads();
    for (int s = 64; s > 0; s >>= 1) { if (d < s) red[d] += red[d + s]; __syncthreads(); }
    float var = red[0] * (1.f / DD);
    float y = dv * rsqrtf(var + 1e-5f) * g[d] + bb[d];
    if (flg && flg[row / flgdiv] > 0.5f) y = xv;
    out[(long)row * DD + d] = y;
}

__global__ __launch_bounds__(128)
void lane_flag_kernel(const float* __restrict__ lm, float* __restrict__ flg, int nb)
{
    int b = blockIdx.x * 128 + threadIdx.x;
    if (b < nb) {
        float s = 0.f;
        for (int j = 0; j < LL; ++j) s += lm[(long)b * LL + j];
        flg[b] = (s == 0.f) ? 1.f : 0.f;
    }
}

__global__ __launch_bounds__(128)
void mean_t_kernel(const float* __restrict__ X, float* __restrict__ out)
{
    int b = blockIdx.x, d = threadIdx.x;
    float s = 0.f;
    for (int t = 0; t < TT; ++t) s += X[((long)b * TT + t) * DD + d];
    out[(long)b * DD + d] = s * (1.f / TT);
}

// nb[b,i,:] = (adj*mask) row-normalized @ lane
__global__ __launch_bounds__(128)
void lane_nb_kernel(const float* __restrict__ adj, const float* __restrict__ mask,
                    const float* __restrict__ lane, float* __restrict__ nb)
{
    __shared__ float am[16][16];
    __shared__ float inv[16];
    int b = blockIdx.x, t = threadIdx.x;
    for (int e = t; e < 256; e += 128) {
        int j = e & 15;
        am[e >> 4][j] = adj[(long)b * 256 + e] * mask[(long)b * LL + j];
    }
    __syncthreads();
    if (t < 16) {
        float s = 0.f;
#pragma unroll
        for (int j = 0; j < 16; ++j) s += am[t][j];
        inv[t] = 1.f / fmaxf(s, 1.f);
    }
    __syncthreads();
    float lv[16];
#pragma unroll
    for (int j = 0; j < 16; ++j) lv[j] = lane[((long)b * LL + j) * DD + t];
#pragma unroll
    for (int i = 0; i < 16; ++i) {
        float s = 0.f;
#pragma unroll
        for (int j = 0; j < 16; ++j) s += am[i][j] * lv[j];
        nb[((long)b * LL + i) * DD + t] = s * inv[i];
    }
}

// hm[b,n,c] = mean_t relu(nh@w + b)
__global__ __launch_bounds__(32)
void nbr1_kernel(const float* __restrict__ nh, const float* __restrict__ w,
                 const float* __restrict__ bb, float* __restrict__ hm)
{
    int bn = blockIdx.x, c = threadIdx.x;
    float w0 = w[c], w1 = w[32 + c], bc = bb[c];
    float acc = 0.f;
    for (int t = 0; t < TT; ++t) {
        float x0 = nh[((long)bn * TT + t) * 2];
        float x1 = nh[((long)bn * TT + t) * 2 + 1];
        acc += fmaxf(x0 * w0 + x1 * w1 + bc, 0.f);
    }
    hm[(long)bn * 32 + c] = acc * (1.f / TT);
}

// ctx[b,j] = masked mean over n of relu(hm@w + b)
__global__ __launch_bounds__(64)
void nbr2_kernel(const float* __restrict__ hm, const float* __restrict__ w,
                 const float* __restrict__ bb, const float* __restrict__ nm,
                 float* __restrict__ ctx)
{
    __shared__ float hs[32 * 32];
    int b = blockIdx.x, j = threadIdx.x;
    for (int e = j; e < 1024; e += 64) hs[e] = hm[(long)b * 1024 + e];
    __syncthreads();
    float s = 0.f, wsum = 0.f;
    for (int n = 0; n < NN; ++n) {
        float m = nm[(long)b * NN + n];
        float dot = bb[j];
        for (int k = 0; k < 32; ++k) dot += hs[n * 32 + k] * w[k * 64 + j];
        s += fmaxf(dot, 0.f) * m;
        wsum += m;
    }
    ctx[(long)b * NBHH + j] = s / fmaxf(wsum, 1.f);
}

// out = constant-velocity + residual
__global__ __launch_bounds__(64)
void final_kernel(const float* __restrict__ sdc, const float* __restrict__ res,
                  float* __restrict__ out)
{
    int b = blockIdx.x, t = threadIdx.x;
    if (t < FUTT * 2) {
        int s = t >> 1, c = t & 1;
        float lp = sdc[((long)b * TT + (TT - 1)) * 2 + c];
        float pv = sdc[((long)b * TT + (TT - 2)) * 2 + c];
        out[(long)b * (FUTT * 2) + t] = lp + (lp - pv) * (float)(s + 1)
                                        + res[(long)b * (FUTT * 2) + t];
    }
}

// =====================================================================
extern "C" void kernel_launch(void* const* d_in, const int* in_sizes, int n_in,
                              void* d_out, int out_size, void* d_ws, size_t ws_size,
                              hipStream_t stream)
{
    (void)in_sizes; (void)n_in; (void)out_size; (void)ws_size;

    const float* sdc   = (const float*)d_in[0];
    const float* lfe   = (const float*)d_in[1];
    const float* lmask = (const float*)d_in[2];
    const float* ladj  = (const float*)d_in[3];
    const float* nh    = (const float*)d_in[4];
    const float* nmask = (const float*)d_in[5];

    // params flattened in JAX pytree order (dict keys sorted alphabetically)
    int ix = 6;
    auto P = [&](void) { return (const float*)d_in[ix++]; };
    const float *ca_bk = P(), *ca_bo = P(), *ca_bq = P(), *ca_bv = P();
    const float *ca_wk = P(), *ca_wo = P(), *ca_wq = P(), *ca_wv = P();
    const float *ca_ln_b = P(), *ca_ln_g = P();
    const float *dec_b1 = P(), *dec_b2 = P(), *dec_w1 = P(), *dec_w2 = P();
    const float *ego_in_b = P(), *ego_in_w = P();
    struct Enc { const float *bk,*bo,*bq,*bv,*ff1_b,*ff1_w,*ff2_b,*ff2_w,
                              *ln1_b,*ln1_g,*ln2_b,*ln2_g,*wk,*wo,*wq,*wv; } enc[2];
    for (int l = 0; l < 2; ++l) {
        enc[l].bk = P(); enc[l].bo = P(); enc[l].bq = P(); enc[l].bv = P();
        enc[l].ff1_b = P(); enc[l].ff1_w = P(); enc[l].ff2_b = P(); enc[l].ff2_w = P();
        enc[l].ln1_b = P(); enc[l].ln1_g = P(); enc[l].ln2_b = P(); enc[l].ln2_g = P();
        enc[l].wk = P(); enc[l].wo = P(); enc[l].wq = P(); enc[l].wv = P();
    }
    const float *ego_pos = P();
    const float *fus_b = P(), *fus_w = P();
    const float *lane_b1 = P(), *lane_b2 = P(), *lane_w1 = P(), *lane_w2 = P();
    const float *mp_b[2], *mp_w[2];
    mp_b[0] = P(); mp_w[0] = P(); mp_b[1] = P(); mp_w[1] = P();
    const float *nbr_emb_b = P(), *nbr_emb_w = P();
    const float *nbr_h_b = P(), *nbr_h_w = P();

    // ---- workspace layout (f32), ~449 MB ----
    float* ws = (float*)d_ws;
    const long BT = (long)BB * TT;      // 90112
    const long BL = (long)BB * LL;      // 131072
    float* X    = ws;
    float* QB   = X    + BT * DD;
    float* KB   = QB   + BT * DD;       // also TL (MP ping-pong)
    float* VB   = KB   + BL * DD;       // also NB / lane-encode tmp
    float* AT   = VB   + BL * DD;
    float* HID  = AT   + BT * DD;       // BT*256; also HM (B*N*32)
    float* LANE = HID  + BT * DFFF;
    float* EGOR = LANE + BL * DD;
    float* FUS  = EGOR + (long)BB * DD;
    float* D1   = FUS  + (long)BB * DD;
    float* RES  = D1   + (long)BB * DD;
    float* CTX  = RES  + (long)BB * (FUTT * 2);
    float* FLG  = CTX  + (long)BB * NBHH;
    float* TL = KB;  float* NB = VB;  float* HM = HID;
    float* outp = (float*)d_out;

    const dim3 blk(256);
    auto gemm = [&](const float* A, int lda, const float* W, int ldw, const float* bias,
                    float* Out, int ldo, int M, int K, int N, int flags, const float* rm) {
        dim3 grid(M / 64, (N + 127) / 128);   // all M here are multiples of 64
        if ((K % 32) || (N % 128))
            wmma_gemm_kernel<true><<<grid, blk, 0, stream>>>(A, lda, W, ldw, bias, Out, ldo,
                                                             M, K, N, flags, rm);
        else
            wmma_gemm_kernel<false><<<grid, blk, 0, stream>>>(A, lda, W, ldw, bias, Out, ldo,
                                                              M, K, N, flags, rm);
    };
    const float scale = 0.17677669529663687f;   // 1/sqrt(32)
    const dim3 attn_grid((BB * NHH) / 8);

    // 1. ego embedding
    embed_kernel<<<dim3((unsigned)BT), dim3(128), 0, stream>>>(sdc, ego_in_w, ego_in_b, ego_pos, X);

    // 2. encoder layers
    for (int l = 0; l < 2; ++l) {
        gemm(X, DD, enc[l].wq, DD, enc[l].bq, QB, DD, (int)BT, DD, DD, 0, nullptr);
        gemm(X, DD, enc[l].wk, DD, enc[l].bk, KB, DD, (int)BT, DD, DD, 0, nullptr);
        gemm(X, DD, enc[l].wv, DD, enc[l].bv, VB, DD, (int)BT, DD, DD, 0, nullptr);
        wmma_attn_kernel<TT, TT, false><<<attn_grid, blk, 0, stream>>>(QB, KB, VB, AT, nullptr, scale);
        gemm(AT, DD, enc[l].wo, DD, enc[l].bo, QB, DD, (int)BT, DD, DD, 0, nullptr);
        add_ln_kernel<<<dim3((unsigned)BT), dim3(128), 0, stream>>>(X, QB, enc[l].ln1_g, enc[l].ln1_b, X, nullptr, 1);
        gemm(X, DD, enc[l].ff1_w, DFFF, enc[l].ff1_b, HID, DFFF, (int)BT, DD, DFFF, GEMM_RELU, nullptr);
        gemm(HID, DFFF, enc[l].ff2_w, DD, enc[l].ff2_b, QB, DD, (int)BT, DFFF, DD, 0, nullptr);
        add_ln_kernel<<<dim3((unsigned)BT), dim3(128), 0, stream>>>(X, QB, enc[l].ln2_g, enc[l].ln2_b, X, nullptr, 1);
    }

    // 3. lane encoder
    gemm(lfe, FF, lane_w1, DD, lane_b1, NB, DD, (int)BL, FF, DD, GEMM_RELU, nullptr);
    gemm(NB, DD, lane_w2, DD, lane_b2, LANE, DD, (int)BL, DD, DD, GEMM_MASK, lmask);

    // 4. message passing (split-K avoids concat): ping LANE -> TL -> LANE
    lane_nb_kernel<<<dim3(BB), dim3(128), 0, stream>>>(ladj, lmask, LANE, NB);
    gemm(LANE, DD, mp_w[0], DD, nullptr, TL, DD, (int)BL, DD, DD, 0, nullptr);
    gemm(NB, DD, mp_w[0] + DD * DD, DD, mp_b[0], TL, DD, (int)BL, DD, DD,
         GEMM_ACCUM | GEMM_RELU | GEMM_MASK, lmask);
    lane_nb_kernel<<<dim3(BB), dim3(128), 0, stream>>>(ladj, lmask, TL, NB);
    gemm(TL, DD, mp_w[1], DD, nullptr, LANE, DD, (int)BL, DD, DD, 0, nullptr);
    gemm(NB, DD, mp_w[1] + DD * DD, DD, mp_b[1], LANE, DD, (int)BL, DD, DD,
         GEMM_ACCUM | GEMM_RELU | GEMM_MASK, lmask);

    // 5. cross attention ego -> lanes
    gemm(X, DD, ca_wq, DD, ca_bq, QB, DD, (int)BT, DD, DD, 0, nullptr);
    gemm(LANE, DD, ca_wk, DD, ca_bk, KB, DD, (int)BL, DD, DD, 0, nullptr);
    gemm(LANE, DD, ca_wv, DD, ca_bv, VB, DD, (int)BL, DD, DD, 0, nullptr);
    wmma_attn_kernel<TT, LL, true><<<attn_grid, blk, 0, stream>>>(QB, KB, VB, AT, lmask, scale);
    gemm(AT, DD, ca_wo, DD, ca_bo, QB, DD, (int)BT, DD, DD, 0, nullptr);
    lane_flag_kernel<<<dim3((BB + 127) / 128), dim3(128), 0, stream>>>(lmask, FLG, BB);
    add_ln_kernel<<<dim3((unsigned)BT), dim3(128), 0, stream>>>(X, QB, ca_ln_g, ca_ln_b, X, FLG, TT);
    mean_t_kernel<<<dim3(BB), dim3(128), 0, stream>>>(X, EGOR);

    // 6. neighbor context
    nbr1_kernel<<<dim3(BB * NN), dim3(32), 0, stream>>>(nh, nbr_emb_w, nbr_emb_b, HM);
    nbr2_kernel<<<dim3(BB), dim3(64), 0, stream>>>(HM, nbr_h_w, nbr_h_b, nmask, CTX);

    // 7. fusion (split-K concat)
    gemm(EGOR, DD, fus_w, DD, nullptr, FUS, DD, BB, DD, DD, 0, nullptr);
    gemm(CTX, NBHH, fus_w + DD * DD, DD, fus_b, FUS, DD, BB, NBHH, DD,
         GEMM_ACCUM | GEMM_RELU, nullptr);

    // 8. decoder
    gemm(FUS, DD, dec_w1, DD, dec_b1, D1, DD, BB, DD, DD, GEMM_RELU, nullptr);
    gemm(D1, DD, dec_w2, FUTT * 2, dec_b2, RES, FUTT * 2, BB, DD, FUTT * 2, 0, nullptr);

    // 9. constant-velocity + residual
    final_kernel<<<dim3(BB), dim3(64), 0, stream>>>(sdc, RES, outp);
}
